// QuantLinear_86363202388342
// MI455X (gfx1250) — matrix operand order repaired and checked by reading, not compiled, not run on hardware
//
#include <hip/hip_runtime.h>

// ---------------------------------------------------------------------------
// QuIP#-style quantized linear for gfx1250 (MI455X), wave32 + WMMA bf16 +
// async global->LDS software pipeline (ASYNCcnt).
//   out = FWHT( (FWHT(x*SU)/64) @ W^T * Wscale ) / 64 * SV + bias
//   W[n,k] = grid[Qidxs[n, k/8]][k%8]      (Wscale folded into decoded W)
// ---------------------------------------------------------------------------

typedef __attribute__((ext_vector_type(16))) __bf16 v16bf;  // 32B: WMMA A/B frag
typedef __attribute__((ext_vector_type(8)))  __bf16 v8bf;   // 16B: one b128
typedef __attribute__((ext_vector_type(8)))  float  v8f;    // WMMA C/D frag

#define RS64 0.015625f   // 1/sqrt(4096)
#define LDA  40          // padded LDS row (80B stride: 16B-aligned, bank-spread)

union ABFrag { v16bf v; v8bf h[2]; };

// CDNA5 async copy: global -> LDS, 16B per lane, tracked with ASYNCcnt.
// SADDR form: 64-bit uniform base in SGPR pair + 32-bit per-lane byte offset.
__device__ __forceinline__ void async_b128(unsigned lds_addr, const void* base,
                                           unsigned byte_off) {
  asm volatile("global_load_async_to_lds_b128 %0, %1, %2"
               :: "v"(lds_addr), "v"(byte_off),
                  "s"((unsigned long long)(uintptr_t)base)
               : "memory");
}

__device__ __forceinline__ void wait_async0() {
#if __has_builtin(__builtin_amdgcn_s_wait_asynccnt)
  __builtin_amdgcn_s_wait_asynccnt(0);
#else
  asm volatile("s_wait_asynccnt 0x0" ::: "memory");
#endif
}

// ---- Kernel 1: Xh = bf16( FWHT(input * SU) / 64 ), one block per token row.
__global__ __launch_bounds__(256) void k_fwht_in(const float* __restrict__ inp,
                                                 const float* __restrict__ SU,
                                                 __bf16* __restrict__ Xh) {
  __shared__ float buf[4096];
  const size_t base = (size_t)blockIdx.x * 4096;
  for (int i = threadIdx.x; i < 4096; i += 256) buf[i] = inp[base + i] * SU[i];
  __syncthreads();
  for (int h = 1; h < 4096; h <<= 1) {
    for (int p = threadIdx.x; p < 2048; p += 256) {
      int i = ((p & ~(h - 1)) << 1) | (p & (h - 1));
      float a = buf[i], b = buf[i + h];
      buf[i]     = a + b;
      buf[i + h] = a - b;
    }
    __syncthreads();
  }
  for (int i = threadIdx.x; i < 4096; i += 256)
    Xh[base + i] = (__bf16)(buf[i] * RS64);
}

// ---- Kernel 2: decode Wd[n][k] = bf16( grid[Qidxs[n, k/8]][k%8] * Wscale[n] )
// One thread per (n, kb): coalesced Qidxs reads, contiguous 16B bf16 stores.
__global__ __launch_bounds__(256) void k_decode(const float* __restrict__ grid,
                                                const int*   __restrict__ Qidxs,
                                                const float* __restrict__ Wscale,
                                                __bf16*      __restrict__ Wd) {
  const int gid = blockIdx.x * 256 + threadIdx.x;   // 4096*512 threads
  const int kb  = gid & 511;
  const int n   = gid >> 9;
  const int idx = Qidxs[(size_t)n * 512 + kb];
  const float ws = Wscale[n];
  const float4* g = (const float4*)(grid + idx * 8);
  float4 g0 = g[0], g1 = g[1];
  v8bf w;
  w[0] = (__bf16)(g0.x * ws); w[1] = (__bf16)(g0.y * ws);
  w[2] = (__bf16)(g0.z * ws); w[3] = (__bf16)(g0.w * ws);
  w[4] = (__bf16)(g1.x * ws); w[5] = (__bf16)(g1.y * ws);
  w[6] = (__bf16)(g1.z * ws); w[7] = (__bf16)(g1.w * ws);
  *(v8bf*)(Wd + (size_t)n * 4096 + kb * 8) = w;
}

// ---- Kernel 3: Y[16384x4096] = Xh @ Wd^T via v_wmma_f32_16x16x32_bf16.
// Block tile 128(M)x128(N), K stepped by 32; 8 waves -> 2x4 grid of 16x16
// accumulators each. Double-buffered LDS fed by global_load_async_to_lds_b128:
// tile k+1 streams into buffer cur^1 while WMMAs consume buffer cur.
__global__ __launch_bounds__(256) void k_gemm(const __bf16* __restrict__ Xh,
                                              const __bf16* __restrict__ Wd,
                                              float* __restrict__ Y) {
  __shared__ __align__(16) __bf16 As[2][128][LDA];
  __shared__ __align__(16) __bf16 Bs[2][128][LDA];

  const int t    = threadIdx.x;
  const int lane = t & 31;          // wave32
  const int wave = t >> 5;          // 8 waves
  const int hi   = lane >> 4;       // lane half selects K-half per ISA layout
  const int l15  = lane & 15;
  const int wm   = (wave >> 1) * 32;   // wave's M offset in block tile
  const int wn   = (wave & 1) * 64;    // wave's N offset in block tile
  const size_t m0 = (size_t)blockIdx.y * 128;
  const size_t n0 = (size_t)blockIdx.x * 128;

  v8f acc[2][4] = {};

  const int r0 = t >> 2;            // staging row within 64-row chunk
  const int c0 = (t & 3) * 8;       // staging col (bf16 units)

  // Issue the 128x32 A and B slabs for K-step `kk` into LDS buffer `buf`:
  // each thread owns 2x16B of A and 2x16B of B (4 async b128 ops).
  auto stage = [&](int buf, int kk) {
#pragma unroll
    for (int it = 0; it < 2; ++it) {
      int r = it * 64 + r0;
      unsigned ga = (unsigned)(((m0 + r) * 4096 + kk + c0) * sizeof(__bf16));
      unsigned gb = (unsigned)(((n0 + r) * 4096 + kk + c0) * sizeof(__bf16));
      async_b128((unsigned)(uintptr_t)&As[buf][r][c0], Xh, ga);
      async_b128((unsigned)(uintptr_t)&Bs[buf][r][c0], Wd, gb);
    }
  };

  // Prologue: fill buffer 0.
  stage(0, 0);
  wait_async0();
  __syncthreads();

  int cur = 0;
  for (int kk = 0; kk < 4096; kk += 32) {
    // Prefetch next K slab into the other buffer (overlaps with WMMAs below).
    if (kk + 32 < 4096) stage(cur ^ 1, kk + 32);

    // A 16x32 bf16 frag: lanes 0-15 row M=l hold K{0..7,16..23},
    // lanes 16-31 hold K{8..15,24..31} (ISA 7.12.2).
    ABFrag a[2], b[4];
#pragma unroll
    for (int i = 0; i < 2; ++i) {
      int r = wm + i * 16 + l15;
      a[i].h[0] = *(const v8bf*)&As[cur][r][hi * 8];
      a[i].h[1] = *(const v8bf*)&As[cur][r][16 + hi * 8];
    }
    // B 32x16 bf16 frag: lanes 0-15 col N=l hold K{0..7,16..23},
    // lanes 16-31 hold K{8..15,24..31} (same K order as A).
#pragma unroll
    for (int j = 0; j < 4; ++j) {
      int r = wn + j * 16 + l15;
      b[j].h[0] = *(const v8bf*)&Bs[cur][r][hi * 8];
      b[j].h[1] = *(const v8bf*)&Bs[cur][r][16 + hi * 8];
    }
#pragma unroll
    for (int i = 0; i < 2; ++i)
#pragma unroll
      for (int j = 0; j < 4; ++j)
        acc[i][j] = __builtin_amdgcn_wmma_f32_16x16x32_bf16(
            false, a[i].v, false, b[j].v, (short)0, acc[i][j], false, false);

    // Publish: my async writes to cur^1 done; everyone's frag reads of cur done
    // (dscnt-waited before the WMMAs above). One barrier per K-step.
    wait_async0();
    __syncthreads();
    cur ^= 1;
  }

  // C/D layout: VGPR v, lanes 0-15 -> row v, lanes 16-31 -> row v+8, col = l15.
#pragma unroll
  for (int i = 0; i < 2; ++i) {
    size_t gm = m0 + wm + i * 16;
#pragma unroll
    for (int j = 0; j < 4; ++j) {
      size_t gn = n0 + wn + j * 16;
#pragma unroll
      for (int v = 0; v < 8; ++v)
        Y[(gm + v + hi * 8) * 4096 + gn + l15] = acc[i][j][v];
    }
  }
}

// ---- Kernel 4: in-place out = FWHT(out)/64 * SV + bias, one block per row.
__global__ __launch_bounds__(256) void k_fwht_out(float* __restrict__ Y,
                                                  const float* __restrict__ SV,
                                                  const float* __restrict__ bias) {
  __shared__ float buf[4096];
  const size_t base = (size_t)blockIdx.x * 4096;
  for (int i = threadIdx.x; i < 4096; i += 256) buf[i] = Y[base + i];
  __syncthreads();
  for (int h = 1; h < 4096; h <<= 1) {
    for (int p = threadIdx.x; p < 2048; p += 256) {
      int i = ((p & ~(h - 1)) << 1) | (p & (h - 1));
      float a = buf[i], b = buf[i + h];
      buf[i]     = a + b;
      buf[i + h] = a - b;
    }
    __syncthreads();
  }
  for (int i = threadIdx.x; i < 4096; i += 256)
    Y[base + i] = buf[i] * RS64 * SV[i] + bias[i];
}

// ---------------------------------------------------------------------------
extern "C" void kernel_launch(void* const* d_in, const int* in_sizes, int n_in,
                              void* d_out, int out_size, void* d_ws, size_t ws_size,
                              hipStream_t stream) {
  const float* inp    = (const float*)d_in[0];   // (8,2048,4096) f32
  const float* SU     = (const float*)d_in[1];   // (4096,) f32
  const float* SV     = (const float*)d_in[2];   // (4096,) f32
  const float* Wscale = (const float*)d_in[3];   // (4096,) f32
  const float* grid   = (const float*)d_in[4];   // (256,8) f32
  const int*   Qidxs  = (const int*)d_in[5];     // (4096,512) i32
  const float* bias   = (const float*)d_in[6];   // (4096,) f32
  float* out = (float*)d_out;                    // (8,2048,4096) f32

  // Workspace layout: bf16 activations (134 MB) + decoded bf16 weights (32 MB).
  __bf16* Xh = (__bf16*)d_ws;
  __bf16* Wd = (__bf16*)((char*)d_ws + (size_t)16384 * 4096 * sizeof(__bf16));

  // 1) Hadamard-rotate + sign-flip activations, quantize to bf16.
  k_fwht_in<<<16384, 256, 0, stream>>>(inp, SU, Xh);

  // 2) Decode codebook weights once (reused by all 16384 tokens), fold Wscale.
  k_decode<<<(4096 * 512) / 256, 256, 0, stream>>>(grid, Qidxs, Wscale, Wd);

  // 3) WMMA bf16 GEMM, async-pipelined: grid (N tiles = 32, M tiles = 128).
  k_gemm<<<dim3(32, 128), 256, 0, stream>>>(Xh, Wd, out);

  // 4) Output Hadamard + SV + bias, in place.
  k_fwht_out<<<16384, 256, 0, stream>>>(out, SV, bias);
}
